// MeshRCNNGraphConvHead_8701603742214
// MI455X (gfx1250) — compile-verified, matrix operand order checked by reading, not run on hardware
//
#include <hip/hip_runtime.h>
#include <hip/hip_bf16.h>

// Problem constants (from reference)
#define NB   64
#define CCH  256
#define HH   14
#define WW   14
#define VV   2562
#define EE   7680
#define HID  128
#define MROWS (NB*VV)            // 163968 = 16 * 10248 (exact multiple of 16)
#define MTILES (MROWS/16)        // 10248

typedef __attribute__((ext_vector_type(16))) __bf16 bf16x16;
typedef __attribute__((ext_vector_type(8)))  __bf16 bf16x8;
typedef __attribute__((ext_vector_type(8)))  float  f32x8;

static __device__ __forceinline__ __bf16 to_bf16(float f) {
  unsigned u = __builtin_bit_cast(unsigned, f);
  unsigned r = u + 0x7FFFu + ((u >> 16) & 1u);   // round-to-nearest-even
  unsigned short h = (unsigned short)(r >> 16);
  return __builtin_bit_cast(__bf16, h);
}

// ---------------------------------------------------------------------------
// 1) vert_align: bilinear sample x(N,C,H,W) at verts -> bf16 A matrix (M x 256)
// ---------------------------------------------------------------------------
__global__ void k_sample(const float* __restrict__ x, const float* __restrict__ verts,
                         __bf16* __restrict__ A) {
  int gid = blockIdx.x * blockDim.x + threadIdx.x;
  int c   = gid & (CCH - 1);
  int row = gid >> 8;
  if (row >= MROWS) return;
  int n = row / VV;
  float gx = verts[(size_t)row * 3 + 0];
  float gy = verts[(size_t)row * 3 + 1];
  float ix = fminf(fmaxf((gx + 1.0f) * 0.5f * (WW - 1), 0.0f), (float)(WW - 1));
  float iy = fminf(fmaxf((gy + 1.0f) * 0.5f * (HH - 1), 0.0f), (float)(HH - 1));
  float x0f = floorf(ix), y0f = floorf(iy);
  float wx = ix - x0f, wy = iy - y0f;
  int x0 = (int)x0f, y0 = (int)y0f;
  int x1 = min(x0 + 1, WW - 1), y1 = min(y0 + 1, HH - 1);
  const float* f = x + ((size_t)n * CCH + c) * (HH * WW);
  float v00 = f[y0 * WW + x0], v01 = f[y0 * WW + x1];
  float v10 = f[y1 * WW + x0], v11 = f[y1 * WW + x1];
  float val = (v00 * (1.f - wx) + v01 * wx) * (1.f - wy) +
              (v10 * (1.f - wx) + v11 * wx) * wy;
  A[(size_t)row * CCH + c] = to_bf16(val);
}

// ---------------------------------------------------------------------------
// 2) Pack fp32 weight (Kact x 128) into per-lane WMMA B fragments, bf16.
//    elem index = ((ntile*nK + ktile)*32 + lane)*16 + e
//    lane: half = lane>>4, col = lane&15; n = ntile*16+col
//    e<8 : k = ktile*32 + half*8 + e ; e>=8 : k = ktile*32 + 16 + half*8 + (e-8)
// ---------------------------------------------------------------------------
__global__ void k_packW(const float* __restrict__ Wm, __bf16* __restrict__ Bp,
                        int Kact, int Kp) {
  int gid = blockIdx.x * blockDim.x + threadIdx.x;
  int nK = Kp >> 5;
  if (gid >= 8 * nK * 512) return;
  int e    = gid & 15;
  int lane = (gid >> 4) & 31;
  int t    = gid >> 9;              // ntile*nK + ktile
  int ntile = t / nK, ktile = t % nK;
  int half = lane >> 4, col = lane & 15;
  int n = ntile * 16 + col;
  int k = ktile * 32 + ((e < 8) ? (half * 8 + e) : (16 + half * 8 + (e - 8)));
  float v = (k < Kact) ? Wm[(size_t)k * HID + n] : 0.0f;
  Bp[(size_t)gid] = to_bf16(v);
}

// ---------------------------------------------------------------------------
// 3) WMMA GEMM, one wave per M-tile; A fragments held in registers and reused
//    across all 8 N-tiles and (optionally) a second B matrix (dual GEMM).
//    KP is compile-time so fragment arrays stay in VGPRs (no scratch).
// ---------------------------------------------------------------------------
template<int KP>
__global__ void __launch_bounds__(256)
k_gemm(const __bf16* __restrict__ A,
       const __bf16* __restrict__ Bp0, const float* __restrict__ bias0,
       float* __restrict__ out0, int relu,
       const __bf16* __restrict__ Bp1, const float* __restrict__ bias1,
       float* __restrict__ out1) {
  constexpr int NK = KP >> 5;
  int lane = threadIdx.x & 31;
  int wave = blockIdx.x * (blockDim.x >> 5) + (threadIdx.x >> 5);
  if (wave >= MTILES) return;            // wave-uniform
  int m_tile = wave;
  int half = lane >> 4, col = lane & 15;

  union Frag { bf16x16 v; bf16x8 h[2]; };

  // Preload all A fragments for this m-tile (NK*8 VGPRs, fully unrolled)
  Frag a[NK];
  const __bf16* arow = A + (size_t)(m_tile * 16 + col) * KP + half * 8;
#pragma unroll
  for (int kt = 0; kt < NK; ++kt) {
    a[kt].h[0] = *(const bf16x8*)(arow + kt * 32);
    a[kt].h[1] = *(const bf16x8*)(arow + kt * 32 + 16);
  }

  int mbase = m_tile * 16 + half * 8;
  for (int pass = 0; pass < 2; ++pass) {
    const __bf16* Bp  = (pass == 0) ? Bp0  : Bp1;
    const float* bias = (pass == 0) ? bias0 : bias1;
    float* out        = (pass == 0) ? out0 : out1;
    if (Bp == nullptr) break;            // uniform
#pragma unroll
    for (int nt = 0; nt < 8; ++nt) {
      float bv = bias[nt * 16 + col];
      f32x8 c = {bv, bv, bv, bv, bv, bv, bv, bv};
      const __bf16* bptr = Bp + ((size_t)nt * NK * 32 + lane) * 16;
#pragma unroll
      for (int kt = 0; kt < NK; ++kt) {
        Frag b;
        b.v = *(const bf16x16*)(bptr + (size_t)kt * 32 * 16);
        c = __builtin_amdgcn_wmma_f32_16x16x32_bf16(
                /*neg_a=*/false, a[kt].v, /*neg_b=*/false, b.v,
                /*c_mod=*/(short)0, c, /*reuse_a=*/false, /*reuse_b=*/false);
      }
      int n = nt * 16 + col;
#pragma unroll
      for (int r = 0; r < 8; ++r) {
        float v = c[r];
        if (relu) v = fmaxf(v, 0.0f);
        out[(size_t)(mbase + r) * HID + n] = v;
      }
    }
  }
}

// ---------------------------------------------------------------------------
// 4) Concat/pack fp32 features -> bf16 A matrix rows [f0 | f1 | verts | 0pad]
// ---------------------------------------------------------------------------
template<int KP>
__global__ void k_pack(const float* __restrict__ f0, int L0,
                       const float* __restrict__ f1, int L1,
                       const float* __restrict__ verts,
                       __bf16* __restrict__ A) {
  int gid = blockIdx.x * blockDim.x + threadIdx.x;
  int k   = gid % KP;
  int row = gid / KP;
  if (row >= MROWS) return;
  float v = 0.0f;
  if (k < L0)                 v = f0[(size_t)row * L0 + k];
  else if (k < L0 + L1)       v = f1[(size_t)row * L1 + (k - L0)];
  else if (k < L0 + L1 + 3)   v = verts[(size_t)row * 3 + (k - L0 - L1)];
  A[(size_t)row * KP + k] = to_bf16(v);
}

// ---------------------------------------------------------------------------
// 5) Edge scatter: undirected neighbor sum via fp32 global atomics into acc.
// ---------------------------------------------------------------------------
__global__ void k_scatter(const float* __restrict__ h1, const int* __restrict__ edges,
                          float* __restrict__ acc) {
  long long gid = (long long)blockIdx.x * blockDim.x + threadIdx.x;
  int f = (int)(gid & (HID - 1));
  long long t = gid >> 7;
  int e = (int)(t % EE);
  int n = (int)(t / EE);
  if (n >= NB) return;
  int i = edges[e * 2 + 0];
  int j = edges[e * 2 + 1];
  size_t base = (size_t)n * VV * HID;
  atomicAdd(&acc[base + (size_t)i * HID + f], h1[base + (size_t)j * HID + f]);
  atomicAdd(&acc[base + (size_t)j * HID + f], h1[base + (size_t)i * HID + f]);
}

// ---------------------------------------------------------------------------
// 6) ReLU(acc) -> nopos fp32, and pack [nopos | verts | 0] bf16 (Kp = 160).
// ---------------------------------------------------------------------------
__global__ void k_relu_pack(const float* __restrict__ acc, float* __restrict__ nopos,
                            const float* __restrict__ verts, __bf16* __restrict__ A) {
  const int Kp = HID + 32;   // 160
  int gid = blockIdx.x * blockDim.x + threadIdx.x;
  int k   = gid % Kp;
  int row = gid / Kp;
  if (row >= MROWS) return;
  float v = 0.0f;
  if (k < HID) {
    v = fmaxf(acc[(size_t)row * HID + k], 0.0f);
    nopos[(size_t)row * HID + k] = v;
  } else if (k < HID + 3) {
    v = verts[(size_t)row * 3 + (k - HID)];
  }
  A[(size_t)row * Kp + k] = to_bf16(v);
}

// ---------------------------------------------------------------------------
// 7) Offset head: deform = tanh([nopos|verts] @ Wo + bo); verts_out = verts + deform
// ---------------------------------------------------------------------------
__global__ void k_offset(const float* __restrict__ nopos, const float* __restrict__ verts_in,
                         const float* __restrict__ Wo, const float* __restrict__ bo,
                         float* __restrict__ verts_out) {
  int row = blockIdx.x * blockDim.x + threadIdx.x;
  if (row >= MROWS) return;
  float a0 = bo[0], a1 = bo[1], a2 = bo[2];
  const float* np = nopos + (size_t)row * HID;
  for (int k = 0; k < HID; ++k) {
    float v = np[k];
    a0 += v * Wo[k * 3 + 0];
    a1 += v * Wo[k * 3 + 1];
    a2 += v * Wo[k * 3 + 2];
  }
  float vx = verts_in[row * 3 + 0], vy = verts_in[row * 3 + 1], vz = verts_in[row * 3 + 2];
  a0 += vx * Wo[128 * 3 + 0] + vy * Wo[129 * 3 + 0] + vz * Wo[130 * 3 + 0];
  a1 += vx * Wo[128 * 3 + 1] + vy * Wo[129 * 3 + 1] + vz * Wo[130 * 3 + 1];
  a2 += vx * Wo[128 * 3 + 2] + vy * Wo[129 * 3 + 2] + vz * Wo[130 * 3 + 2];
  verts_out[row * 3 + 0] = vx + tanhf(a0);
  verts_out[row * 3 + 1] = vy + tanhf(a1);
  verts_out[row * 3 + 2] = vz + tanhf(a2);
}

// ---------------------------------------------------------------------------
extern "C" void kernel_launch(void* const* d_in, const int* in_sizes, int n_in,
                              void* d_out, int out_size, void* d_ws, size_t ws_size,
                              hipStream_t stream) {
  (void)in_sizes; (void)n_in; (void)out_size; (void)ws_size;
  const float* x      = (const float*)d_in[0];
  const float* verts0 = (const float*)d_in[1];
  const int*   edges  = (const int*)d_in[2];

  char* ws = (char*)d_ws;
  size_t off = 0;
  auto carve = [&](size_t bytes) -> char* {
    char* p = ws + off;
    off += (bytes + 255) & ~(size_t)255;
    return p;
  };
  __bf16* Abuf = (__bf16*)carve((size_t)MROWS * 288 * sizeof(__bf16));
  float*  img  = (float*)carve((size_t)MROWS * HID * sizeof(float));
  float*  h0   = (float*)carve((size_t)MROWS * HID * sizeof(float));
  float*  h1   = (float*)carve((size_t)MROWS * HID * sizeof(float));
  float*  nopos= (float*)carve((size_t)MROWS * HID * sizeof(float));
  __bf16* Bp0  = (__bf16*)carve((size_t)288 * 128 * sizeof(__bf16));
  __bf16* Bp1  = (__bf16*)carve((size_t)288 * 128 * sizeof(__bf16));
  float*  vA   = (float*)carve((size_t)MROWS * 3 * sizeof(float));
  float*  vB   = (float*)carve((size_t)MROWS * 3 * sizeof(float));

  auto packW = [&](const float* Wm, int Kact, int Kp, __bf16* Bp) {
    int threads = 8 * (Kp >> 5) * 512;
    k_packW<<<dim3((threads + 255) / 256), dim3(256), 0, stream>>>(Wm, Bp, Kact, Kp);
  };
  // Dual-output GEMM: second B may be nullptr. 8 waves/block, one m-tile/wave.
  auto gemm = [&](const __bf16* B0, const float* b0, float* o0, int relu,
                  const __bf16* B1, const float* b1, float* o1, int Kp) {
    dim3 g((MTILES + 7) / 8), blk(256);
    switch (Kp) {
      case 160: k_gemm<160><<<g, blk, 0, stream>>>(Abuf, B0, b0, o0, relu, B1, b1, o1); break;
      case 256: k_gemm<256><<<g, blk, 0, stream>>>(Abuf, B0, b0, o0, relu, B1, b1, o1); break;
      case 288: k_gemm<288><<<g, blk, 0, stream>>>(Abuf, B0, b0, o0, relu, B1, b1, o1); break;
      default: break;
    }
  };
  auto pack = [&](const float* f0, int L0, const float* f1, int L1,
                  const float* v, int Kp) {
    long long threads = (long long)MROWS * Kp;
    dim3 g((unsigned)((threads + 255) / 256)), blk(256);
    if (Kp == 160) k_pack<160><<<g, blk, 0, stream>>>(f0, L0, f1, L1, v, Abuf);
    else           k_pack<288><<<g, blk, 0, stream>>>(f0, L0, f1, L1, v, Abuf);
  };

  const float* verts_cur = verts0;
  float* verts_next[3] = { vA, vB, (float*)d_out };

  for (int s = 0; s < 3; ++s) {
    int base = 3 + s * 16;
    const float* bnW  = (const float*)d_in[base + 0];
    const float* bnB  = (const float*)d_in[base + 1];
    const float* offW = (const float*)d_in[base + 14];
    const float* offB = (const float*)d_in[base + 15];

    // vert_align -> bf16 A (M x 256)
    k_sample<<<dim3(MROWS), dim3(256), 0, stream>>>(x, verts_cur, Abuf);

    // bottleneck GEMM, ReLU -> img
    packW(bnW, 256, 256, Bp0);
    gemm(Bp0, bnB, img, 1, nullptr, nullptr, nullptr, 256);

    // pack vf for gconv0
    int din, Kp;
    if (s == 0) {
      din = HID + 3; Kp = 160;
      pack(img, HID, nullptr, 0, verts_cur, Kp);
    } else {
      din = 2 * HID + 3; Kp = 288;
      pack(nopos, HID, img, HID, verts_cur, Kp);
    }

    for (int g = 0; g < 3; ++g) {
      const float* w0W = (const float*)d_in[base + 2 + g * 4 + 0];
      const float* w0b = (const float*)d_in[base + 2 + g * 4 + 1];
      const float* w1W = (const float*)d_in[base + 2 + g * 4 + 2];
      const float* w1b = (const float*)d_in[base + 2 + g * 4 + 3];
      packW(w0W, din, Kp, Bp0);
      packW(w1W, din, Kp, Bp1);
      // fused dual GEMM: h0 = vf@w0+b0, h1 = vf@w1+b1 (A fragments reused)
      gemm(Bp0, w0b, h0, 0, Bp1, w1b, h1, Kp);
      {
        long long threads = (long long)NB * EE * HID;
        k_scatter<<<dim3((unsigned)((threads + 255) / 256)), dim3(256), 0, stream>>>(
            h1, edges, h0);
      }
      {
        long long threads = (long long)MROWS * 160;
        k_relu_pack<<<dim3((unsigned)((threads + 255) / 256)), dim3(256), 0, stream>>>(
            h0, nopos, verts_cur, Abuf);
      }
      din = HID + 3; Kp = 160;
    }

    k_offset<<<dim3((MROWS + 255) / 256), dim3(256), 0, stream>>>(
        nopos, verts_cur, offW, offB, verts_next[s]);
    verts_cur = verts_next[s];
  }
}
